// MPNNPredictor_66494683677213
// MI455X (gfx1250) — matrix operand order repaired and checked by reading, not compiled, not run on hardware
//
#include <hip/hip_runtime.h>

// ---------------------------------------------------------------------------
// MI455X (gfx1250) implementation of the MPNN predictor.
//   * All GEMMs: v_wmma_f32_16x16x32_bf16, f32 accumulate. Each wave computes
//     a 16x64 output strip; all 4 B fragments are fetched before the WMMA
//     chain so waits retire progressively instead of fully draining loads.
//   * The 410MB bf16 per-edge weight tensor `we` (read 6x -> roofline
//     bottleneck) is streamed into LDS by the Tensor Data Mover:
//     one tensor_load_to_lds per wave (8KB block), s_wait_tensorcnt, then
//     bank-conflict-free LDS compute + f32 global atomics for segment-sum.
//   * GRU fused per 16-row tile: relu+bias -> bf16 LDS tiles -> 24 WMMA jobs
//     across 8 waves -> gate math -> in-place hidden update.
// ---------------------------------------------------------------------------

typedef __attribute__((ext_vector_type(16))) __bf16        v16bf;
typedef __attribute__((ext_vector_type(8)))  float         v8f;
typedef __attribute__((ext_vector_type(4)))  unsigned int  u32x4;
typedef __attribute__((ext_vector_type(8)))  unsigned int  u32x8;

union ABfrag { v16bf v; uint4 q[2]; };

__device__ __forceinline__ unsigned short f2bf(float f) {
    unsigned u = __float_as_uint(f);
    u += 0x7FFFu + ((u >> 16) & 1u);          // round-to-nearest-even
    return (unsigned short)(u >> 16);
}

// ---------------- f32 -> bf16 conversion with optional zero K-padding -------
__global__ void cvt_bf16(const float* __restrict__ src,
                         unsigned short* __restrict__ dst,
                         int rows, int scols, int dcols) {
    long long i = (long long)blockIdx.x * 256 + threadIdx.x;
    long long tot = (long long)rows * dcols;
    if (i >= tot) return;
    int r = (int)(i / dcols), c = (int)(i % dcols);
    dst[i] = (c < scols) ? f2bf(src[(long long)r * scols + c]) : (unsigned short)0;
}

// ---------------- bf16 WMMA GEMM: C = relu?(A @ W^T + bias) -----------------
// A: (M,K) bf16 row-major.  W: (N,K) bf16 row-major (torch-style).
// One wave per 16x64 output strip: 4 accumulators share each A fragment.
// M multiple of 16, N multiple of 64, K multiple of 32.
__global__ __launch_bounds__(32)
void gemm_bf16(const unsigned short* __restrict__ A,
               const unsigned short* __restrict__ W,
               const float* __restrict__ bias,
               float* __restrict__ Cf,            // f32 output (or null)
               unsigned short* __restrict__ Cb,   // bf16 output (or null)
               int N, int K, int relu) {
    const int lane = threadIdx.x & 31;
    const int half = lane >> 4;
    const int lm   = lane & 15;
    const int mt = blockIdx.x, nb = blockIdx.y;   // nb: 64-column block
    const int arow = mt * 16 + lm;                // A-fragment row for this lane

    v8f acc[4] = {{}, {}, {}, {}};
    for (int k0 = 0; k0 < K; k0 += 32) {
        // Fetch A fragment + all 4 B fragments up front, then run the WMMA
        // chain: waits retire progressively (loadcnt 6/4/2/0) instead of a
        // full drain before every WMMA.
        ABfrag a, b[4];
        const unsigned short* ap = A + (size_t)arow * K + k0 + half * 16;
        a.q[0] = *(const uint4*)(ap);             // K = k0+half*16  .. +7
        a.q[1] = *(const uint4*)(ap + 8);         // K = k0+half*16+8.. +15
#pragma unroll
        for (int t = 0; t < 4; ++t) {
            const unsigned short* bp =
                W + (size_t)(nb * 64 + t * 16 + lm) * K + k0 + half * 16;
            b[t].q[0] = *(const uint4*)(bp);      // K = k0+half*16  .. +7
            b[t].q[1] = *(const uint4*)(bp + 8);  // K = k0+half*16+8.. +15
        }
#pragma unroll
        for (int t = 0; t < 4; ++t) {
            acc[t] = __builtin_amdgcn_wmma_f32_16x16x32_bf16(
                         false, a.v, false, b[t].v, (short)0, acc[t], false, false);
        }
    }
#pragma unroll
    for (int t = 0; t < 4; ++t) {
        const int col = nb * 64 + t * 16 + lm;
        const float bv = bias ? bias[col] : 0.0f;
#pragma unroll
        for (int r = 0; r < 8; ++r) {
            float v = acc[t][r] + bv;
            if (relu) v = fmaxf(v, 0.0f);
            const int orow = mt * 16 + half * 8 + r;
            if (Cf) Cf[(size_t)orow * N + col] = v;
            else    Cb[(size_t)orow * N + col] = f2bf(v);
        }
    }
}

// ---------------- message passing: m = einsum('ei,eio->eo'); scatter-add ----
// One wave per edge. The wave's 8KB bf16 weight block is DMA'd into LDS by
// the Tensor Data Mover (tensor_load_to_lds, TENSORcnt), then the matvec is
// computed from LDS (consecutive lanes -> consecutive banks).
__global__ __launch_bounds__(256)
void msg_scatter(const float* __restrict__ h,
                 const unsigned short* __restrict__ we,   // (E,64,64) bf16
                 const int* __restrict__ src,
                 const int* __restrict__ dst,
                 float* __restrict__ neigh, int E) {
    __shared__ float          hs[8][64];
    __shared__ unsigned short wt[8][4096];        // 8 x 8KB weight tiles
    const int lane = threadIdx.x;                 // 0..31
    const int wv   = threadIdx.y;                 // 0..7
    const int e    = blockIdx.x * 8 + wv;
    const bool act = (e < E);

    // ---- Tensor DMA descriptor (2-group form, VADDR2/3 = NULL) ----
    // 2D tensor: 1 row x 4096 elems of 2 bytes; tile = whole row (8KB).
    const unsigned long long gaddr =
        (unsigned long long)(const void*)(we + (size_t)(act ? e : 0) * 4096);
    const unsigned lds_base =
        (unsigned)(unsigned long long)(const void*)&wt[wv][0];  // LDS byte offset
    u32x4 g0;
    g0[0] = 0x1u;                                  // count=1, user mode
    g0[1] = lds_base;                              // lds_addr (bytes)
    g0[2] = (unsigned)(gaddr & 0xFFFFFFFFu);       // global_addr[31:0]
    g0[3] = (unsigned)((gaddr >> 32) & 0x1FFFFFFu) // global_addr[56:32]
          | (2u << 30);                            // type=2 ("image")
    u32x8 g1;
    g1[0] = 1u << 16;                              // data_size=1 (2B), wg_mask=0
    g1[1] = 0x1000u << 16;                         // tensor_dim0 = 4096 (lo16)
    g1[2] = 1u << 16;                              // tensor_dim0 hi=0; tensor_dim1=1
    g1[3] = 0x1000u << 16;                         // tensor_dim1 hi=0; tile_dim0=4096
    g1[4] = 1u;                                    // tile_dim1=1, tile_dim2=0
    g1[5] = 4096u;                                 // tensor_dim0_stride (lo32)
    g1[6] = 0u;                                    // stride hi; dim1_stride lo
    g1[7] = 0u;
    asm volatile("tensor_load_to_lds %0, %1" :: "s"(g0), "s"(g1) : "memory");

    int s = 0, d = 0;
    if (act) {
        s = src[e]; d = dst[e];
        hs[wv][lane]      = h[(size_t)s * 64 + lane];
        hs[wv][lane + 32] = h[(size_t)s * 64 + lane + 32];
    }
    __builtin_amdgcn_s_wait_tensorcnt(0);          // our 8KB tile is in LDS
    __syncthreads();
    if (!act) return;

    float a0 = 0.0f, a1 = 0.0f;
    const unsigned short* wp = &wt[wv][lane * 2];
#pragma unroll 4
    for (int i = 0; i < 64; ++i) {
        const float hv = hs[wv][i];
        const unsigned u = *(const unsigned*)(wp + i * 64);
        a0 += hv * __uint_as_float(u << 16);          // out feature 2*lane
        a1 += hv * __uint_as_float(u & 0xFFFF0000u);  // out feature 2*lane+1
    }
    atomicAdd(&neigh[(size_t)d * 64 + lane * 2],     a0);
    atomicAdd(&neigh[(size_t)d * 64 + lane * 2 + 1], a1);
}

// ---------------- fused relu + GRU cell over a 16-node tile -----------------
__global__ __launch_bounds__(256)
void gru_fused(const float* __restrict__ neigh,
               float* __restrict__ hidden,               // in/out (V,64)
               const float* __restrict__ conv_b,
               const unsigned short* __restrict__ wih,   // (192,64) bf16
               const unsigned short* __restrict__ whh,   // (192,64) bf16
               const float* __restrict__ bih,
               const float* __restrict__ bhh) {
    __shared__ unsigned short xbf[16][64];
    __shared__ unsigned short hbf[16][64];
    __shared__ float gi[16][192];
    __shared__ float gh[16][192];

    const int tid = threadIdx.x;
    const int m0  = blockIdx.x * 16;

    // Stage 1: x = relu(neigh + conv_b); bf16 tiles of x and hidden into LDS
#pragma unroll
    for (int p = 0; p < 4; ++p) {
        const int idx = p * 256 + tid;
        const int r = idx >> 6, c = idx & 63;
        const size_t off = (size_t)(m0 + r) * 64 + c;
        const float xv = fmaxf(neigh[off] + conv_b[c], 0.0f);
        xbf[r][c] = f2bf(xv);
        hbf[r][c] = f2bf(hidden[off]);
    }
    __syncthreads();

    // Stage 2: 24 WMMA tile-jobs (12 gi N-tiles + 12 gh N-tiles) over 8 waves
    const int wv = tid >> 5, lane = tid & 31, half = lane >> 4, lm = lane & 15;
    for (int job = wv; job < 24; job += 8) {
        const bool isGi = (job < 12);
        const unsigned short (*Asrc)[64] = isGi ? xbf : hbf;
        const unsigned short* Wt = isGi ? wih : whh;
        const int nt = job % 12;
        const int n  = nt * 16 + lm;
        v8f acc = {};
#pragma unroll
        for (int k0 = 0; k0 < 64; k0 += 32) {
            ABfrag a, b;
            const unsigned short* ap = &Asrc[lm][k0 + half * 16];
            a.q[0] = *(const uint4*)(ap);         // ds_load_b128
            a.q[1] = *(const uint4*)(ap + 8);
            const unsigned short* bp = Wt + (size_t)n * 64 + k0 + half * 16;
            b.q[0] = *(const uint4*)(bp);
            b.q[1] = *(const uint4*)(bp + 8);
            acc = __builtin_amdgcn_wmma_f32_16x16x32_bf16(
                      false, a.v, false, b.v, (short)0, acc, false, false);
        }
        const float bv = (isGi ? bih : bhh)[n];
        float (*G)[192] = isGi ? gi : gh;
#pragma unroll
        for (int r = 0; r < 8; ++r) G[half * 8 + r][n] = acc[r] + bv;
    }
    __syncthreads();

    // Stage 3: gates (torch order r,z,n) and in-place hidden update
#pragma unroll
    for (int p = 0; p < 4; ++p) {
        const int idx = p * 256 + tid;
        const int r = idx >> 6, o = idx & 63;
        const float rg = 1.0f / (1.0f + __expf(-(gi[r][o]       + gh[r][o])));
        const float zg = 1.0f / (1.0f + __expf(-(gi[r][64 + o]  + gh[r][64 + o])));
        const float ng = tanhf(gi[r][128 + o] + rg * gh[r][128 + o]);
        const size_t off = (size_t)(m0 + r) * 64 + o;
        hidden[off] = (1.0f - zg) * ng + zg * hidden[off];
    }
}

// ---------------- per-edge dot products (u_dot_v) ---------------------------
__global__ __launch_bounds__(256)
void edge_dot(const float* __restrict__ h,
              const int* __restrict__ sa, const int* __restrict__ da,
              float* __restrict__ out, int E) {
    const int lane = threadIdx.x, wv = threadIdx.y;
    const int e = blockIdx.x * 8 + wv;
    if (e >= E) return;
    const int s = sa[e], d = da[e];
    float p = h[(size_t)s * 64 + lane]      * h[(size_t)d * 64 + lane]
            + h[(size_t)s * 64 + lane + 32] * h[(size_t)d * 64 + lane + 32];
#pragma unroll
    for (int off = 16; off; off >>= 1) p += __shfl_xor(p, off, 32);
    if (lane == 0) out[e] = p;
}

// ---------------------------------------------------------------------------
extern "C" void kernel_launch(void* const* d_in, const int* in_sizes, int n_in,
                              void* d_out, int out_size, void* d_ws, size_t ws_size,
                              hipStream_t stream) {
    (void)in_sizes; (void)n_in; (void)out_size; (void)ws_size;
    const int V = 20000, E = 50000;

    const float* node_feats = (const float*)d_in[0];
    const float* edge_feats = (const float*)d_in[1];
    const float* proj_w     = (const float*)d_in[2];
    const float* proj_b     = (const float*)d_in[3];
    const float* en_w1      = (const float*)d_in[4];
    const float* en_b1      = (const float*)d_in[5];
    const float* en_w2      = (const float*)d_in[6];
    const float* en_b2      = (const float*)d_in[7];
    const float* conv_b     = (const float*)d_in[8];
    const float* w_ih       = (const float*)d_in[9];
    const float* w_hh       = (const float*)d_in[10];
    const float* b_ih       = (const float*)d_in[11];
    const float* b_hh       = (const float*)d_in[12];
    const int*   src        = (const int*)d_in[13];
    const int*   dst        = (const int*)d_in[14];
    const int*   nsrc       = (const int*)d_in[15];
    const int*   ndst       = (const int*)d_in[16];
    float* out = (float*)d_out;

    // ----- workspace carve-up (256B aligned) -----
    char* ws = (char*)d_ws;
    size_t off = 0;
    auto carve = [&](size_t bytes) -> char* {
        char* p = ws + off;
        off = (off + bytes + 255) & ~(size_t)255;
        return p;
    };
    unsigned short* nf_bf  = (unsigned short*)carve((size_t)V * 128 * 2);
    unsigned short* pw_bf  = (unsigned short*)carve((size_t)64 * 128 * 2);
    unsigned short* ef_bf  = (unsigned short*)carve((size_t)E * 32 * 2);   // K 16->32 pad
    unsigned short* w1_bf  = (unsigned short*)carve((size_t)128 * 32 * 2); // K 16->32 pad
    unsigned short* w2_bf  = (unsigned short*)carve((size_t)4096 * 128 * 2);
    unsigned short* wih_bf = (unsigned short*)carve((size_t)192 * 64 * 2);
    unsigned short* whh_bf = (unsigned short*)carve((size_t)192 * 64 * 2);
    unsigned short* t_bf   = (unsigned short*)carve((size_t)E * 128 * 2);
    float*          hidden = (float*)carve((size_t)V * 64 * 4);
    float*          neigh  = (float*)carve((size_t)V * 64 * 4);
    unsigned short* we_bf  = (unsigned short*)carve((size_t)E * 4096 * 2);

    // ----- one-time bf16 conversions (zero-padded where K<32) -----
    auto cvt = [&](const float* s, unsigned short* d, int rows, int sc, int dc) {
        long long tot = (long long)rows * dc;
        int g = (int)((tot + 255) / 256);
        hipLaunchKernelGGL(cvt_bf16, dim3(g), dim3(256), 0, stream, s, d, rows, sc, dc);
    };
    cvt(node_feats, nf_bf, V, 128, 128);
    cvt(proj_w,     pw_bf, 64, 128, 128);
    cvt(edge_feats, ef_bf, E, 16, 32);
    cvt(en_w1,      w1_bf, 128, 16, 32);
    cvt(en_w2,      w2_bf, 4096, 128, 128);
    cvt(w_ih,       wih_bf, 192, 64, 64);
    cvt(w_hh,       whh_bf, 192, 64, 64);

    // ----- node projection: hidden = relu(nf @ proj_w^T + b)  (M=V,N=64,K=128)
    hipLaunchKernelGGL(gemm_bf16, dim3(V / 16, 64 / 64), dim3(32), 0, stream,
                       nf_bf, pw_bf, proj_b, hidden, (unsigned short*)nullptr,
                       64, 128, 1);

    // ----- edge network layer 1: t = relu(ef @ w1^T + b1)  (M=E,N=128,K=32pad)
    hipLaunchKernelGGL(gemm_bf16, dim3(E / 16, 128 / 64), dim3(32), 0, stream,
                       ef_bf, w1_bf, en_b1, (float*)nullptr, t_bf,
                       128, 32, 1);

    // ----- edge network layer 2: we = t @ w2^T + b2  (M=E,N=4096,K=128), 52 GF
    hipLaunchKernelGGL(gemm_bf16, dim3(E / 16, 4096 / 64), dim3(32), 0, stream,
                       t_bf, w2_bf, en_b2, (float*)nullptr, we_bf,
                       4096, 128, 0);

    // ----- 6 message-passing + GRU steps -----
    for (int step = 0; step < 6; ++step) {
        hipMemsetAsync(neigh, 0, (size_t)V * 64 * 4, stream);
        hipLaunchKernelGGL(msg_scatter, dim3((E + 7) / 8), dim3(32, 8), 0, stream,
                           hidden, we_bf, src, dst, neigh, E);
        hipLaunchKernelGGL(gru_fused, dim3(V / 16), dim3(256), 0, stream,
                           neigh, hidden, conv_b, wih_bf, whh_bf, b_ih, b_hh);
    }

    // ----- edge scores: pos then neg -----
    hipLaunchKernelGGL(edge_dot, dim3((E + 7) / 8), dim3(32, 8), 0, stream,
                       hidden, src, dst, out, E);
    hipLaunchKernelGGL(edge_dot, dim3((E + 7) / 8), dim3(32, 8), 0, stream,
                       hidden, nsrc, ndst, out + E, E);
}
